// ContrastiveModel_85976655332046
// MI455X (gfx1250) — matrix operand order repaired and checked by reading, not compiled, not run on hardware
//
#include <hip/hip_runtime.h>
#include <hip/hip_bf16.h>

#define BB 32
#define SS 512
#define DD 768

typedef __attribute__((ext_vector_type(16))) _Float16 v16h;
typedef __attribute__((ext_vector_type(8)))  _Float16 v8h;
typedef __attribute__((ext_vector_type(4)))  _Float16 v4h;
typedef __attribute__((ext_vector_type(8)))  float    v8f;

#define NEG_INF __int_as_float((int)0xFF800000)

// Order-independent float atomic max via sign-split int atomics.
// Buffers are initialized to -inf (0xFF800000): int view is very negative,
// unsigned view is the max among ordered negative floats -> both paths safe.
__device__ __forceinline__ void atomicMaxFloat(float* addr, float val) {
  if (val >= 0.0f) atomicMax((int*)addr, __float_as_int(val));
  else             atomicMin((unsigned int*)addr, __float_as_uint(val));
}

// ---------------------------------------------------------------------------
// Kernel 1: L2-normalize each token embedding (fp32 -> f16), one wave per row.
// Also initializes rowmax/colmax to -inf (runs before the GEMM on the stream).
// ---------------------------------------------------------------------------
__global__ __launch_bounds__(256) void normalize_kernel(
    const float* __restrict__ e1, const float* __restrict__ e2,
    _Float16* __restrict__ n1, _Float16* __restrict__ n2,
    float* __restrict__ rowmax, float* __restrict__ colmax)
{
  const int gid  = blockIdx.x * 256 + threadIdx.x;
  const int wave = gid >> 5;           // one wave32 per row; 2*B*S rows total
  const int lane = gid & 31;
  const bool second = wave >= BB * SS;
  const int row = second ? (wave - BB * SS) : wave;

  const float* src = (second ? e2 : e1) + (size_t)row * DD;
  _Float16*    dst = (second ? n2 : n1) + (size_t)row * DD;

  float4 vals[6];
  float ss = 0.0f;
#pragma unroll
  for (int c = 0; c < 6; ++c) {        // 6 x float4 per lane = 24 floats
    float4 v = ((const float4*)src)[lane + 32 * c];
    vals[c] = v;
    ss += v.x * v.x + v.y * v.y + v.z * v.z + v.w * v.w;
  }
#pragma unroll
  for (int off = 16; off > 0; off >>= 1) ss += __shfl_xor(ss, off, 32);

  const float inv = 1.0f / fmaxf(sqrtf(ss), 1e-8f);
#pragma unroll
  for (int c = 0; c < 6; ++c) {
    v4h o;
    o[0] = (_Float16)(vals[c].x * inv);
    o[1] = (_Float16)(vals[c].y * inv);
    o[2] = (_Float16)(vals[c].z * inv);
    o[3] = (_Float16)(vals[c].w * inv);
    ((v4h*)dst)[lane + 32 * c] = o;
  }
  if (lane == 0) (second ? colmax : rowmax)[row] = NEG_INF;
}

// ---------------------------------------------------------------------------
// Kernel 2: one wave computes a 32(i) x 64(j) block of sim = n1 . n2^T via
// 24 K-steps x 8 v_wmma_f32_16x16x32_f16 (2 i-tiles x 4 j-tiles register
// blocking). B tiles are loaded as a batch ahead of the 8-WMMA burst and the
// K-loop is unrolled x2 so next-step loads overlap this step's WMMAs (the
// grid is small -> occupancy is grid-limited, latency must hide in-wave).
//
// A layout (16x32, this code's K order, not the ISA table's): lane = {h=lane/16, m=lane%16}; per-lane
// halves [0..15] = row m, K = k0+16h .. +15 contiguous (same order as B).
// B layout (K x N): lane holds column n = lane%16, K = k0+16h .. +15
// contiguous -> exactly 16 contiguous halves of n2 row (j-col), since
// B[k][n] = n2[j0+n][k].
// ---------------------------------------------------------------------------
__global__ __launch_bounds__(32) void sim_tile_kernel(
    const _Float16* __restrict__ n1, const _Float16* __restrict__ n2,
    const int* __restrict__ mask1, const int* __restrict__ mask2,
    float* __restrict__ rowmax, float* __restrict__ colmax)
{
  const int lane = threadIdx.x;
  const int m = lane & 15;
  const int h = lane >> 4;
  const int i0 = blockIdx.x * 32;
  const int j0 = blockIdx.y * 64;
  const int b  = blockIdx.z;

  const _Float16* arow[2];
#pragma unroll
  for (int it = 0; it < 2; ++it)
    arow[it] = n1 + (size_t)(b * SS + i0 + it * 16 + m) * DD + 16 * h;
  const _Float16* brow[4];
#pragma unroll
  for (int jt = 0; jt < 4; ++jt)
    brow[jt] = n2 + (size_t)(b * SS + j0 + jt * 16 + m) * DD + 16 * h;

  v8f acc[2][4] = {};

#pragma unroll 2
  for (int k0 = 0; k0 < DD; k0 += 32) {
    // Batch-issue all loads for this K-step, then run the 8-WMMA burst.
    v16h bt[4];
#pragma unroll
    for (int jt = 0; jt < 4; ++jt)
      bt[jt] = *(const v16h*)(brow[jt] + k0);      // 16 contiguous K-halves
    v16h a[2];
#pragma unroll
    for (int it = 0; it < 2; ++it) {
      a[it] = *(const v16h*)(arow[it] + k0);       // K = k0+16h .. +15
    }
#pragma unroll
    for (int jt = 0; jt < 4; ++jt) {
#pragma unroll
      for (int it = 0; it < 2; ++it) {
        acc[it][jt] = __builtin_amdgcn_wmma_f32_16x16x32_f16(
            /*neg_a=*/false, a[it], /*neg_b=*/false, bt[jt],
            /*c_mod=*/(short)0, acc[it][jt],
            /*reuse_a=*/false, /*reuse_b=*/false);
      }
    }
  }

  // 16 row-validity bits per i-tile (lanes 16..31 duplicate lanes 0..15)
  unsigned rowbits[2];
#pragma unroll
  for (int it = 0; it < 2; ++it)
    rowbits[it] =
        (unsigned)__ballot(mask1[b * SS + i0 + it * 16 + m] != 0) & 0xFFFFu;

  // C/D layout: lane holds column n = lane%16; VGPR r holds row M = r + 8h.
#pragma unroll
  for (int jt = 0; jt < 4; ++jt) {
    const int jcol = j0 + jt * 16 + m;
    const bool colvalid = mask2[b * SS + jcol] != 0;
    float colv = NEG_INF;
#pragma unroll
    for (int it = 0; it < 2; ++it) {
#pragma unroll
      for (int r = 0; r < 8; ++r) {
        const float v = acc[it][jt][r];
        // --- row max over this tile's 16 columns (mask invalid columns) ---
        float vr = colvalid ? v : NEG_INF;
        vr = fmaxf(vr, __shfl_xor(vr, 1, 16));
        vr = fmaxf(vr, __shfl_xor(vr, 2, 16));
        vr = fmaxf(vr, __shfl_xor(vr, 4, 16));
        vr = fmaxf(vr, __shfl_xor(vr, 8, 16));
        if (m == 0)
          atomicMaxFloat(&rowmax[b * SS + i0 + it * 16 + r + 8 * h], vr);
        // --- column max over this tile's 16 rows (mask invalid rows) ---
        const bool rowvalid = (rowbits[it] >> (r + 8 * h)) & 1u;
        colv = fmaxf(colv, rowvalid ? v : NEG_INF);
      }
    }
    colv = fmaxf(colv, __shfl_xor(colv, 16, 32));   // combine the two halves
    if (h == 0) atomicMaxFloat(&colmax[b * SS + jcol], colv);
  }
}

// ---------------------------------------------------------------------------
// Kernel 3: per-batch score = (sum row_max[m1] + sum col_max[m2]) / max(n1+n2,1)
// ---------------------------------------------------------------------------
__global__ __launch_bounds__(256) void finalize_kernel(
    const float* __restrict__ rowmax, const float* __restrict__ colmax,
    const int* __restrict__ mask1, const int* __restrict__ mask2,
    float* __restrict__ out)
{
  const int b = blockIdx.x;
  const int t = threadIdx.x;
  float sum = 0.0f, cnt = 0.0f;
  for (int s = t; s < SS; s += 256) {
    const int m1v = mask1[b * SS + s];
    const int m2v = mask2[b * SS + s];
    sum += (m1v ? rowmax[b * SS + s] : 0.0f) +
           (m2v ? colmax[b * SS + s] : 0.0f);
    cnt += (float)(m1v + m2v);
  }
#pragma unroll
  for (int off = 16; off > 0; off >>= 1) {
    sum += __shfl_xor(sum, off, 32);
    cnt += __shfl_xor(cnt, off, 32);
  }
  __shared__ float ssum[8], scnt[8];
  const int wid = t >> 5, lane = t & 31;
  if (lane == 0) { ssum[wid] = sum; scnt[wid] = cnt; }
  __syncthreads();
  if (t == 0) {
    float s_ = 0.0f, c_ = 0.0f;
    for (int w = 0; w < 8; ++w) { s_ += ssum[w]; c_ += scnt[w]; }
    out[b] = s_ / fmaxf(c_, 1.0f);
  }
}

extern "C" void kernel_launch(void* const* d_in, const int* in_sizes, int n_in,
                              void* d_out, int out_size, void* d_ws, size_t ws_size,
                              hipStream_t stream) {
  (void)in_sizes; (void)n_in; (void)out_size; (void)ws_size;
  const float* emb1  = (const float*)d_in[0];
  const float* emb2  = (const float*)d_in[1];
  const int*   mask1 = (const int*)d_in[2];
  const int*   mask2 = (const int*)d_in[3];
  float* out = (float*)d_out;

  char* ws = (char*)d_ws;
  const size_t f16buf = (size_t)BB * SS * DD * sizeof(_Float16);  // 25.2 MB
  _Float16* n1 = (_Float16*)ws;
  _Float16* n2 = (_Float16*)(ws + f16buf);
  float* rowmax = (float*)(ws + 2 * f16buf);
  float* colmax = rowmax + BB * SS;

  // 1) normalize + cast to f16 + init max buffers (one wave per row)
  normalize_kernel<<<(2 * BB * SS) / 8, 256, 0, stream>>>(
      emb1, emb2, n1, n2, rowmax, colmax);

  // 2) WMMA tiles + masked row/col max: grid (S/32) x (S/64) x B, wave32 blocks
  dim3 g2(SS / 32, SS / 64, BB);
  sim_tile_kernel<<<g2, 32, 0, stream>>>(n1, n2, mask1, mask2, rowmax, colmax);

  // 3) per-batch score
  finalize_kernel<<<BB, 256, 0, stream>>>(rowmax, colmax, mask1, mask2, out);
}
